// RAVENNTM_23175643529544
// MI455X (gfx1250) — matrix operand order repaired and checked by reading, not compiled, not run on hardware
//
#include <hip/hip_runtime.h>
#include <hip/hip_bf16.h>

typedef __attribute__((ext_vector_type(16))) _Float16     v16h;
typedef __attribute__((ext_vector_type(8)))  float        v8f;
typedef __attribute__((ext_vector_type(4)))  unsigned int u32x4;
typedef __attribute__((ext_vector_type(8)))  int          i32x8;
typedef __attribute__((ext_vector_type(4)))  int          i32x4;

#define TSTR 36   // LDS tile row pitch in halves: 32 data halves (16 DW) + 2 pad DW

// ---------------------------------------------------------------------------
// Implicit-GEMM convolution, fused BN (+residual) (+ReLU), f16 in / f16 out,
// f32 accumulate via v_wmma_f32_16x16x32_f16.  Tile: BM=64 x BN=128 x BK=32.
// 256 threads = 8 waves in 4x2; each wave computes 16x64 of C -> 4 WMMAs/Kstep
// with one reused A fragment.  All geometry is compile-time so the im2col and
// epilogue index math lowers to constant-divisor mul/shift (no div expansions).
// Weight tile is fetched by the Tensor Data Mover (tensor_load_to_lds) with
// LDS row padding 16DW+2DW so the TDM output pitch matches TSTR.
// ---------------------------------------------------------------------------
template <int CIN, int H, int W, int COUT, int OH, int OW,
          int KH, int KW, int STRIDE, int PAD>
__global__ __launch_bounds__(256) void conv_wmma_kernel(
    const _Float16* __restrict__ in,   // [64, CIN, H, W]
    const _Float16* __restrict__ wgt,  // [COUT, KTOT] f16 row-major
    _Float16* __restrict__ out,        // [64, COUT, OH, OW]
    const _Float16* __restrict__ res,  // residual or null
    const float* __restrict__ bn_scale, const float* __restrict__ bn_bias,
    const float* __restrict__ bn_mean,  const float* __restrict__ bn_var,
    int relu) {
  constexpr int KTOT = CIN * KH * KW;
  constexpr int KPAD = (KTOT + 31) & ~31;
  constexpr int NTOT = 64 * OH * OW;
  constexpr int KHW  = KH * KW;
  constexpr int OHW  = OH * OW;

  __shared__ __align__(16) _Float16 Ash[64 * TSTR];    // weights [m][k]
  __shared__ __align__(16) _Float16 Bsh[128 * TSTR];   // im2col  [n][k]

  const int tid  = threadIdx.x;
  const int lane = tid & 31;
  const int wave = tid >> 5;       // 0..7
  const int wr   = wave >> 1;      // m offset 16*wr
  const int wc   = wave & 1;       // n offset 64*wc
  const int l15  = lane & 15;
  const int lhf  = lane >> 4;

  const int n_base = blockIdx.x * 128;
  const int m_base = blockIdx.y * 64;

  const unsigned lds_A = (unsigned)(unsigned long long)(void*)&Ash[0];

  if (tid == 0)  // warm L2 for this block's weight rows (global_prefetch_b8)
    __builtin_prefetch(wgt + (unsigned long long)m_base * KTOT, 0, 1);

  v8f acc[4] = {v8f{}, v8f{}, v8f{}, v8f{}};

  for (int k0 = 0; k0 < KPAD; k0 += 32) {
    __syncthreads();   // previous tiles fully consumed

    // ---- weight tile via Tensor Data Mover (wave 0; TENSORcnt) ----
    if (wave == 0) {
      unsigned long long gaddr = (unsigned long long)(const void*)wgt
          + ((unsigned long long)m_base * KTOT + (unsigned long long)k0) * 2ull;
      unsigned td0 = (unsigned)(KTOT - k0);   // remaining K (TDM OOB -> 0)
      unsigned td1 = (unsigned)(COUT - m_base);

      u32x4 g0;
      g0[0] = 1u;                                           // count=1
      g0[1] = lds_A;                                        // LDS byte address
      g0[2] = (unsigned)(gaddr & 0xFFFFFFFFull);
      g0[3] = (unsigned)((gaddr >> 32) & 0x1FFFFFFull) | (2u << 30);  // addr hi | type=2

      i32x8 g1;
      g1[0] = (int)((1u << 16) |   // data_size = 2 bytes
                    (1u << 20) |   // pad_enable
                    (3u << 22) |   // pad_interval = 16 DWORDs (one 64B row)
                    (1u << 25));   // pad_amount   = 2 DWORDs -> pitch 72B = TSTR
      g1[1] = (int)((td0 & 0xFFFFu) << 16);
      g1[2] = (int)(((td0 >> 16) & 0xFFFFu) | ((td1 & 0xFFFFu) << 16));
      g1[3] = (int)(((td1 >> 16) & 0xFFFFu) | (32u << 16));   // tile_dim0=32
      g1[4] = (int)(64u);                                     // tile_dim1=64
      g1[5] = (int)(unsigned)KTOT;                            // dim0_stride lo
      g1[6] = 0;                                              // dim0_stride hi
      g1[7] = 0;
      i32x4 gz  = {0, 0, 0, 0};
      i32x8 gz8 = {0, 0, 0, 0, 0, 0, 0, 0};
      __builtin_amdgcn_tensor_load_to_lds(g0, g1, gz, gz, gz8, 0);
    }

    // ---- im2col B tile gather: 128x32 halves, 16 per thread ----
#pragma unroll
    for (int i = 0; i < 16; ++i) {
      int f   = tid * 16 + i;
      int n_l = f >> 5;
      int k_l = f & 31;
      int k   = k0 + k_l;
      int n   = n_base + n_l;
      _Float16 v = (_Float16)0.0f;
      if (k < KTOT && n < NTOT) {
        int img = n / OHW; int rem = n - img * OHW;     // constant divisors
        int oy  = rem / OW; int ox = rem - oy * OW;
        int ci  = k / KHW;  int kr = k - ci * KHW;
        int ky  = kr / KW;  int kx = kr - ky * KW;
        int iy  = oy * STRIDE - PAD + ky;
        int ix  = ox * STRIDE - PAD + kx;
        if (iy >= 0 && iy < H && ix >= 0 && ix < W)
          v = in[(((unsigned long long)img * CIN + ci) * H + iy) * W + ix];
      }
      Bsh[n_l * TSTR + k_l] = v;
    }

    if (wave == 0) __builtin_amdgcn_s_wait_tensorcnt(0);
    __syncthreads();   // A (TDM) and B (VALU) tiles visible

    // ---- fragments per ISA 7.12.2 layouts ----
    const _Float16* Arow = &Ash[(wr * 16 + l15) * TSTR];
    v16h a;
#pragma unroll
    for (int e = 0; e < 8; ++e) a[e] = Arow[lhf * 8 + e];           // K = h*8+e
#pragma unroll
    for (int e = 0; e < 8; ++e) a[8 + e] = Arow[16 + lhf * 8 + e];  // K = 16+h*8+e

#pragma unroll
    for (int q = 0; q < 4; ++q) {
      const _Float16* Brow = &Bsh[(wc * 64 + q * 16 + l15) * TSTR];
      v16h b;
#pragma unroll
      for (int e = 0; e < 16; ++e) b[e] = Brow[(e < 8) ? (lhf * 8 + e) : (16 + lhf * 8 + (e - 8))];  // K = h*8+e | 16+h*8+(e-8)
      acc[q] = __builtin_amdgcn_wmma_f32_16x16x32_f16(
          false, a, false, b, (short)0, acc[q], false, false);
    }
  }

  // ---- epilogue: BN (+res) (+ReLU), store f16; C layout: m=8h+j, n=l%16 ----
  float scv[8], biv[8];
#pragma unroll
  for (int j = 0; j < 8; ++j) {
    int m = m_base + wr * 16 + lhf * 8 + j;     // COUT is a multiple of 64
    float inv = rsqrtf(bn_var[m] + 1e-5f);
    scv[j] = bn_scale[m] * inv;
    biv[j] = bn_bias[m] - bn_mean[m] * scv[j];
  }
#pragma unroll
  for (int q = 0; q < 4; ++q) {
    int n = n_base + wc * 64 + q * 16 + l15;
    if (n >= NTOT) continue;
    int img = n / OHW; int rem = n - img * OHW;
    int oy  = rem / OW; int ox = rem - oy * OW;
#pragma unroll
    for (int j = 0; j < 8; ++j) {
      int m = m_base + wr * 16 + lhf * 8 + j;
      float val = acc[q][j] * scv[j] + biv[j];
      unsigned long long oidx =
          (((unsigned long long)img * COUT + m) * OH + oy) * OW + ox;
      if (res)  val += (float)res[oidx];
      if (relu) val = fmaxf(val, 0.0f);
      out[oidx] = (_Float16)val;
    }
  }
}

// ---------------------------------------------------------------------------
// Small support kernels
// ---------------------------------------------------------------------------
__global__ void cvt_f32_f16_kernel(const float* __restrict__ s, _Float16* __restrict__ d, int n) {
  int i = blockIdx.x * blockDim.x + threadIdx.x;
  if (i < n) d[i] = (_Float16)s[i];
}

__global__ void maxpool_kernel(const _Float16* __restrict__ in, _Float16* __restrict__ out) {
  // [64,64,112,112] -> [64,64,56,56], 3x3 stride2 pad1
  long long idx = (long long)blockIdx.x * blockDim.x + threadIdx.x;
  const long long tot = 64ll * 64 * 56 * 56;
  if (idx >= tot) return;
  int ox = (int)(idx % 56); long long t1 = idx / 56;
  int oy = (int)(t1 % 56);  long long t2 = t1 / 56;
  int ch = (int)(t2 % 64);  int img = (int)(t2 / 64);
  const _Float16* base = in + (((long long)img * 64 + ch) * 112) * 112;
  float m = -3.0e38f;
  for (int ky = 0; ky < 3; ++ky) {
    int iy = oy * 2 - 1 + ky;
    if (iy < 0 || iy >= 112) continue;
    for (int kx = 0; kx < 3; ++kx) {
      int ix = ox * 2 - 1 + kx;
      if (ix < 0 || ix >= 112) continue;
      m = fmaxf(m, (float)base[iy * 112 + ix]);
    }
  }
  out[idx] = (_Float16)m;
}

__global__ void avgpool_kernel(const _Float16* __restrict__ in, float* __restrict__ out) {
  int idx = blockIdx.x * blockDim.x + threadIdx.x;   // 64*512 (img,c)
  if (idx >= 64 * 512) return;
  const _Float16* p = in + (long long)idx * 49;
  float s = 0.f;
#pragma unroll
  for (int i = 0; i < 49; ++i) s += (float)p[i];
  out[idx] = s * (1.0f / 49.0f);
}

__global__ void ctxnorm_kernel(const float* __restrict__ f, const float* __restrict__ gamma,
                               const float* __restrict__ beta, float* __restrict__ o) {
  int idx = blockIdx.x * blockDim.x + threadIdx.x;   // 4*512 (b,d)
  if (idx >= 4 * 512) return;
  int b = idx >> 9, d = idx & 511;
  float mu = 0.f;
  for (int t = 0; t < 16; ++t) mu += f[(b * 16 + t) * 512 + d];
  mu *= (1.0f / 16.0f);
  float v = 0.f;
  for (int t = 0; t < 16; ++t) { float x = f[(b * 16 + t) * 512 + d] - mu; v += x * x; }
  float sg = sqrtf(v * (1.0f / 15.0f) + 1e-8f);
  for (int t = 0; t < 16; ++t)
    o[(b * 16 + t) * 512 + d] = (f[(b * 16 + t) * 512 + d] - mu) / sg * gamma[d] + beta[d];
}

// ---------------------------------------------------------------------------
// MANN: one persistent block (32 waves). 16 sequential steps; state in LDS,
// external memory matrix M in workspace. ~70 MFLOP total: latency-bound, VALU.
// ---------------------------------------------------------------------------
struct MannParams {
  const float* z;       // [64][512] (img = b*16 + t), context-normed
  const float* M0;      // [20][256]
  const float* W_if;    // [256][768]
  const float* W_out;   // [512][38]
  const float* Wh;      // [256][1024]
  const float* Wx;      // [768][1024]
  const float* b_if; const float* b_lstm; const float* b_out;
  float* M;             // ws [4][20][256]
  float* outp;          // d_out, 152 floats
};

__device__ __forceinline__ float sigf(float x) { return 1.0f / (1.0f + expf(-x)); }

__global__ __launch_bounds__(1024) void mann_kernel(MannParams p) {
  __shared__ float h[4][256], c[4][256], r[4][256];
  __shared__ float gates[4][1024];
  __shared__ float iface[4][768];
  __shared__ float simr[4][20], simw[4][20];
  __shared__ float wr[4][4], ww[4][4];
  __shared__ int   ir[4][4], iw[4][4];
  const int tid = threadIdx.x;
  const int b8  = tid >> 8, j8 = tid & 255;

  h[b8][j8] = 0.f; c[b8][j8] = 0.f; r[b8][j8] = 0.f;
  for (int idx = tid; idx < 4 * 20 * 256; idx += 1024) p.M[idx] = p.M0[idx % (20 * 256)];
  __syncthreads();

  for (int t = 0; t < 16; ++t) {
    // gates[b][j] = [z_t, r] @ Wx + h @ Wh + b_lstm
    for (int b = 0; b < 4; ++b) {
      const float* zz = p.z + (long long)(b * 16 + t) * 512;
      float g = p.b_lstm[tid];
      for (int k = 0; k < 512; ++k) g += zz[k]   * p.Wx[k * 1024 + tid];
      for (int k = 0; k < 256; ++k) g += r[b][k] * p.Wx[(512 + k) * 1024 + tid];
      for (int k = 0; k < 256; ++k) g += h[b][k] * p.Wh[k * 1024 + tid];
      gates[b][tid] = g;
    }
    __syncthreads();
    {
      float ig = sigf(gates[b8][j8]),        fg = sigf(gates[b8][256 + j8]);
      float gg = tanhf(gates[b8][512 + j8]), og = sigf(gates[b8][768 + j8]);
      float cn = fg * c[b8][j8] + ig * gg;
      c[b8][j8] = cn;
      h[b8][j8] = og * tanhf(cn);
    }
    __syncthreads();
    for (int idx = tid; idx < 4 * 768; idx += 1024) {   // interface: rk|wk|wv
      int b = idx / 768, jj = idx - b * 768;
      float v = p.b_if[jj];
      for (int k = 0; k < 256; ++k) v += h[b][k] * p.W_if[k * 768 + jj];
      iface[b][jj] = v;
    }
    __syncthreads();
    for (int idx = tid; idx < 4 * 20 * 2; idx += 1024) { // cosine similarities
      int head = idx / 80; int rem = idx - head * 80;
      int b = rem / 20, s = rem - b * 20;
      const float* key  = &iface[b][head * 256];
      const float* mrow = p.M + ((long long)b * 20 + s) * 256;
      float dot = 0.f, nk = 0.f, nm = 0.f;
      for (int k = 0; k < 256; ++k) {
        float kv = key[k], mv = mrow[k];
        dot += kv * mv; nk += kv * kv; nm += mv * mv;
      }
      float sim = dot / ((sqrtf(nk) + 1e-8f) * (sqrtf(nm) + 1e-8f));
      if (head) simw[b][s] = sim; else simr[b][s] = sim;
    }
    __syncthreads();
    if (tid < 8) {                                       // top-4 + softmax
      int b = tid & 3, head = tid >> 2;
      float* sim = head ? simw[b] : simr[b];
      float vv[4]; int ii[4]; bool used[20];
      for (int s = 0; s < 20; ++s) used[s] = false;
      for (int k = 0; k < 4; ++k) {
        float best = -3.0e38f; int bi = 0;
        for (int s = 0; s < 20; ++s)
          if (!used[s] && sim[s] > best) { best = sim[s]; bi = s; }
        used[bi] = true; vv[k] = best; ii[k] = bi;
      }
      float mx = vv[0], den = 0.f;
      for (int k = 0; k < 4; ++k) { vv[k] = expf(vv[k] - mx); den += vv[k]; }
      for (int k = 0; k < 4; ++k) {
        if (head) { ww[b][k] = vv[k] / den; iw[b][k] = ii[k]; }
        else      { wr[b][k] = vv[k] / den; ir[b][k] = ii[k]; }
      }
    }
    __syncthreads();
    {   // read head then write head; thread (b,d) exclusively owns column d
      float rv = 0.f;
      for (int k = 0; k < 4; ++k)
        rv += wr[b8][k] * p.M[((long long)b8 * 20 + ir[b8][k]) * 256 + j8];
      float wv = tanhf(iface[b8][512 + j8]);
      for (int k = 0; k < 4; ++k)
        p.M[((long long)b8 * 20 + iw[b8][k]) * 256 + j8] += ww[b8][k] * wv;
      r[b8][j8] = rv;
    }
    __syncthreads();
  }

  if (tid < 4 * 38) {   // out = [h|r] @ W_out + b_out, split 8|9|21, b-major
    int b = tid / 38, o = tid - b * 38;
    float v = p.b_out[o];
    for (int k = 0; k < 256; ++k) v += h[b][k] * p.W_out[k * 38 + o];
    for (int k = 0; k < 256; ++k) v += r[b][k] * p.W_out[(256 + k) * 38 + o];
    int pos;
    if (o < 8)       pos = b * 8 + o;
    else if (o < 17) pos = 32 + b * 9 + (o - 8);
    else             pos = 68 + b * 21 + (o - 17);
    p.outp[pos] = v;
  }
}

// ---------------------------------------------------------------------------
// Host orchestration
// ---------------------------------------------------------------------------
struct BNP { const float *bias, *mean, *scale, *var; };
struct BlockP { BNP bn1, bn2; const float *conv1, *conv2, *down_conv; BNP down_bn; bool has_down; };

template <int CIN, int H, int W, int COUT, int OH, int OW,
          int KH, int KW, int ST, int PD>
static void launch_conv(hipStream_t stream, const _Float16* in, const _Float16* w,
                        _Float16* out, const _Float16* res, const BNP& bn, int relu) {
  constexpr int NTOT = 64 * OH * OW;
  dim3 grid((unsigned)((NTOT + 127) / 128), (unsigned)(COUT / 64));
  conv_wmma_kernel<CIN, H, W, COUT, OH, OW, KH, KW, ST, PD>
      <<<grid, dim3(256), 0, stream>>>(in, w, out, res,
                                       bn.scale, bn.bias, bn.mean, bn.var, relu);
}

extern "C" void kernel_launch(void* const* d_in, const int* in_sizes, int n_in,
                              void* d_out, int out_size, void* d_ws, size_t ws_size,
                              hipStream_t stream) {
  (void)in_sizes; (void)n_in; (void)out_size; (void)ws_size;

  // ---- unpack inputs (jax tree-flatten order: dict keys sorted) ----
  int ii = 0;
  auto nextf = [&]() { return (const float*)d_in[ii++]; };
  const float* x = nextf();
  nextf();  // embedding (unused)
  nextf();  // indicator (unused)
  auto readBN = [&]() { BNP b; b.bias = nextf(); b.mean = nextf(); b.scale = nextf(); b.var = nextf(); return b; };
  BNP   bn1c  = readBN();
  const float* conv1w = nextf();
  auto readBlock = [&](bool down) {
    BlockP p; p.bn1 = readBN(); p.bn2 = readBN();
    p.conv1 = nextf(); p.conv2 = nextf(); p.has_down = down; p.down_conv = nullptr;
    if (down) { p.down_bn = readBN(); p.down_conv = nextf(); }
    return p;
  };
  BlockP L[4][2];
  L[0][0] = readBlock(false); L[0][1] = readBlock(false);
  L[1][0] = readBlock(true);  L[1][1] = readBlock(false);
  L[2][0] = readBlock(true);  L[2][1] = readBlock(false);
  L[3][0] = readBlock(true);  L[3][1] = readBlock(false);
  // mann_params sorted: M0, W_if, W_out, Wh, Wx, b_if, b_lstm, b_out
  const float* M0   = nextf(); const float* W_if = nextf(); const float* W_out = nextf();
  const float* Wh   = nextf(); const float* Wx   = nextf(); const float* b_if  = nextf();
  const float* b_ls = nextf(); const float* b_o  = nextf();
  const float* gamma = nextf(); const float* beta = nextf();

  // ---- workspace layout ----
  char* ws = (char*)d_ws; size_t off = 0;
  auto walloc = [&](size_t bytes) { off = (off + 255) & ~(size_t)255; void* p = ws + off; off += bytes; return p; };
  _Float16* actA = (_Float16*)walloc(64ull * 64 * 112 * 112 * 2);  // 102.8 MB
  _Float16* actB = (_Float16*)walloc(64ull * 64 * 112 * 112 * 2);  // 102.8 MB
  _Float16* actS = (_Float16*)walloc(64ull * 64 * 56 * 56 * 2);    // 25.7 MB
  _Float16* wf16 = (_Float16*)walloc(12ull * 1024 * 1024 * 2);     // 24 MB
  float* feats = (float*)walloc(64ull * 512 * 4);
  float* featn = (float*)walloc(64ull * 512 * 4);
  float* Mbuf  = (float*)walloc(4ull * 20 * 256 * 4);

  auto cvt = [&](const float* src, _Float16* dst, long long n) {
    cvt_f32_f16_kernel<<<dim3((unsigned)((n + 255) / 256)), dim3(256), 0, stream>>>(src, dst, (int)n);
  };
  size_t woff = 0;
  auto cvtW = [&](const float* src, long long n) {
    _Float16* dst = wf16 + woff; woff += (size_t)n;
    cvt(src, dst, n);
    return (const _Float16*)dst;
  };

  // ---- stem ----
  cvt(x, actA, 64ll * 224 * 224);
  launch_conv<1, 224, 224, 64, 112, 112, 7, 7, 2, 3>(
      stream, actA, cvtW(conv1w, 64ll * 49), actB, nullptr, bn1c, 1);
  {
    long long tot = 64ll * 64 * 56 * 56;
    maxpool_kernel<<<dim3((unsigned)((tot + 255) / 256)), dim3(256), 0, stream>>>(actB, actA);
  }

  // ---- layer1 (64ch, 56x56): two identity blocks ----
  launch_conv<64, 56, 56, 64, 56, 56, 3, 3, 1, 1>(
      stream, actA, cvtW(L[0][0].conv1, 36864), actB, nullptr, L[0][0].bn1, 1);
  launch_conv<64, 56, 56, 64, 56, 56, 3, 3, 1, 1>(
      stream, actB, cvtW(L[0][0].conv2, 36864), actS, actA, L[0][0].bn2, 1);
  launch_conv<64, 56, 56, 64, 56, 56, 3, 3, 1, 1>(
      stream, actS, cvtW(L[0][1].conv1, 36864), actB, nullptr, L[0][1].bn1, 1);
  launch_conv<64, 56, 56, 64, 56, 56, 3, 3, 1, 1>(
      stream, actB, cvtW(L[0][1].conv2, 36864), actA, actS, L[0][1].bn2, 1);

  // ---- layer2 (128ch, 28x28) ----
  launch_conv<64, 56, 56, 128, 28, 28, 1, 1, 2, 0>(
      stream, actA, cvtW(L[1][0].down_conv, 8192), actS, nullptr, L[1][0].down_bn, 0);
  launch_conv<64, 56, 56, 128, 28, 28, 3, 3, 2, 1>(
      stream, actA, cvtW(L[1][0].conv1, 73728), actB, nullptr, L[1][0].bn1, 1);
  launch_conv<128, 28, 28, 128, 28, 28, 3, 3, 1, 1>(
      stream, actB, cvtW(L[1][0].conv2, 147456), actA, actS, L[1][0].bn2, 1);
  launch_conv<128, 28, 28, 128, 28, 28, 3, 3, 1, 1>(
      stream, actA, cvtW(L[1][1].conv1, 147456), actB, nullptr, L[1][1].bn1, 1);
  launch_conv<128, 28, 28, 128, 28, 28, 3, 3, 1, 1>(
      stream, actB, cvtW(L[1][1].conv2, 147456), actS, actA, L[1][1].bn2, 1);

  // ---- layer3 (256ch, 14x14) ----
  launch_conv<128, 28, 28, 256, 14, 14, 1, 1, 2, 0>(
      stream, actS, cvtW(L[2][0].down_conv, 32768), actA, nullptr, L[2][0].down_bn, 0);
  launch_conv<128, 28, 28, 256, 14, 14, 3, 3, 2, 1>(
      stream, actS, cvtW(L[2][0].conv1, 294912), actB, nullptr, L[2][0].bn1, 1);
  launch_conv<256, 14, 14, 256, 14, 14, 3, 3, 1, 1>(
      stream, actB, cvtW(L[2][0].conv2, 589824), actS, actA, L[2][0].bn2, 1);
  launch_conv<256, 14, 14, 256, 14, 14, 3, 3, 1, 1>(
      stream, actS, cvtW(L[2][1].conv1, 589824), actB, nullptr, L[2][1].bn1, 1);
  launch_conv<256, 14, 14, 256, 14, 14, 3, 3, 1, 1>(
      stream, actB, cvtW(L[2][1].conv2, 589824), actA, actS, L[2][1].bn2, 1);

  // ---- layer4 (512ch, 7x7) ----
  launch_conv<256, 14, 14, 512, 7, 7, 1, 1, 2, 0>(
      stream, actA, cvtW(L[3][0].down_conv, 131072), actS, nullptr, L[3][0].down_bn, 0);
  launch_conv<256, 14, 14, 512, 7, 7, 3, 3, 2, 1>(
      stream, actA, cvtW(L[3][0].conv1, 1179648), actB, nullptr, L[3][0].bn1, 1);
  launch_conv<512, 7, 7, 512, 7, 7, 3, 3, 1, 1>(
      stream, actB, cvtW(L[3][0].conv2, 2359296), actA, actS, L[3][0].bn2, 1);
  launch_conv<512, 7, 7, 512, 7, 7, 3, 3, 1, 1>(
      stream, actA, cvtW(L[3][1].conv1, 2359296), actB, nullptr, L[3][1].bn1, 1);
  launch_conv<512, 7, 7, 512, 7, 7, 3, 3, 1, 1>(
      stream, actB, cvtW(L[3][1].conv2, 2359296), actS, actA, L[3][1].bn2, 1);

  // ---- head ----
  avgpool_kernel<<<dim3(128), dim3(256), 0, stream>>>(actS, feats);
  ctxnorm_kernel<<<dim3(8), dim3(256), 0, stream>>>(feats, gamma, beta, featn);

  MannParams mp;
  mp.z = featn; mp.M0 = M0; mp.W_if = W_if; mp.W_out = W_out; mp.Wh = Wh; mp.Wx = Wx;
  mp.b_if = b_if; mp.b_lstm = b_ls; mp.b_out = b_o; mp.M = Mbuf; mp.outp = (float*)d_out;
  mann_kernel<<<dim3(1), dim3(1024), 0, stream>>>(mp);
}
